// STDPConv2d_22170621182019
// MI455X (gfx1250) — matrix operand order repaired and checked by reading, not compiled, not run on hardware
//
#include <hip/hip_runtime.h>
#include <hip/hip_bf16.h>

typedef __attribute__((ext_vector_type(16))) __bf16 v16bf;
typedef __attribute__((ext_vector_type(8)))  float  v8f;

#define NTOT  (64*64*64*64)        // 16,777,216 elements per (B,64,64,64) array
#define CHW   (64*64*64)           // 262,144 elements per (64,64,64) array
#define NW    (9*64*64)            // 36,864 weight elements

__device__ __forceinline__ unsigned short f2bf(float f) {
    // round-to-nearest-even f32 -> bf16
    unsigned int u = __float_as_uint(f);
    u += 0x7FFFu + ((u >> 16) & 1u);
    return (unsigned short)(u >> 16);
}

__device__ __forceinline__ float clip01(float v) {
    return fminf(fmaxf(v, 0.0f), 1.0f);
}

union Frag { v16bf v; uint4 q[2]; };

// ---------------------------------------------------------------------------
// Kernel 0: one-shot weight prep: f32 [cout][cin][tap] -> bf16 [tap][cout][cin]
// so conv blocks can stage weights into LDS as a linear async byte copy.
// ---------------------------------------------------------------------------
__global__ __launch_bounds__(256) void snn_wprep_kernel(
    const float* __restrict__ weights, unsigned short* __restrict__ wbf)
{
    int i = blockIdx.x * 256 + threadIdx.x;            // 0..NW-1
    int cin = i & 63, cout = (i >> 6) & 63, tap = i >> 12;
    wbf[(tap*64 + cout)*64 + cin] = f2bf(weights[(cout*64 + cin)*9 + tap]);
}

// ---------------------------------------------------------------------------
// Kernel 1: fused im2col-free WMMA conv (bf16 x bf16 -> f32) + LIF + new_post.
// One block = one (batch b, output row y): 64 c_out x 64 pixels.
// 512 threads = 16 waves; wave (mt,nt) computes a 16x16 (c_out x pixel) tile.
// Weights arrive via GLOBAL_LOAD_ASYNC_TO_LDS_B128 (overlapped with input
// staging), inputs are converted f32->bf16 in-flight.
// ---------------------------------------------------------------------------
__global__ __launch_bounds__(512) void snn_conv_lif_kernel(
    const float* __restrict__ in_spikes, const unsigned short* __restrict__ wbf,
    const float* __restrict__ membrane,  const float* __restrict__ thresholds,
    const float* __restrict__ trace_post,
    float* __restrict__ out_spk, float* __restrict__ out_mem,
    float* __restrict__ out_th,  float* __restrict__ out_post)
{
    __shared__ __align__(16) unsigned short wlds[NW];        // [tap][cout][cin]  72 KB
    __shared__ __align__(16) unsigned short xlds[3*66*64];   // [r][x+1][cin]     25 KB

    const int tid = threadIdx.x;
    const int b   = blockIdx.x >> 6;
    const int y   = blockIdx.x & 63;

    // ---- kick off async weight image copy (linear: LDS layout == global layout)
#if __has_builtin(__builtin_amdgcn_global_load_async_to_lds_b128)
    {
        typedef int v4i_vs __attribute__((vector_size(16)));
        v4i_vs* gflat = (v4i_vs*)const_cast<unsigned short*>(wbf);
        __attribute__((address_space(1))) v4i_vs* gsrc =
            (__attribute__((address_space(1))) v4i_vs*)gflat;
        __attribute__((address_space(3))) v4i_vs* ldst =
            (__attribute__((address_space(3))) v4i_vs*)wlds;
        for (int i = tid; i < (NW*2)/16; i += 512)
            __builtin_amdgcn_global_load_async_to_lds_b128(gsrc + i, ldst + i, 0, 0);
    }
#else
    for (int i = tid * 8; i < NW; i += 512 * 8)
        *reinterpret_cast<uint4*>(&wlds[i]) = *reinterpret_cast<const uint4*>(&wbf[i]);
#endif

    // ---- stage input rows y-1..y+1 as bf16 (zero-padded halo), c_in fastest
    for (int i = tid; i < 3*64*64; i += 512) {
        int xx = i & 63, c = (i >> 6) & 63, r = i >> 12;
        int yy = y + r - 1;
        float v = (yy >= 0 && yy < 64)
                  ? in_spikes[(((b*64 + c)*64) + yy)*64 + xx] : 0.0f;
        xlds[(r*66 + xx + 1)*64 + c] = f2bf(v);
    }
    if (tid < 3*64*2) {   // halo columns x=-1 and x=64 are always padding
        int side = tid & 1, c = (tid >> 1) & 63, r = tid >> 7;
        xlds[(r*66 + (side ? 65 : 0))*64 + c] = 0;
    }

#if __has_builtin(__builtin_amdgcn_s_wait_asynccnt)
    __builtin_amdgcn_s_wait_asynccnt(0);
#else
    asm volatile("s_wait_asynccnt 0x0" ::: "memory");
#endif
    __syncthreads();

    const int lane  = tid & 31;
    const int wave  = tid >> 5;
    const int mbase = (wave >> 2) * 16;   // c_out tile base
    const int x0    = (wave & 3) * 16;    // pixel tile base
    const int hi    = lane >> 4;          // lane half (ISA wave32 fragment layout)
    const int nn    = lane & 15;

    v8f acc = {};
    #pragma unroll
    for (int tap = 0; tap < 9; ++tap) {
        const int ky = tap / 3, kx = tap % 3;
        // B: input column for this lane's pixel, shifted by (kx-1): LDS x-index = px + kx
        const unsigned short* brow = &xlds[(ky*66 + x0 + nn + kx)*64];
        // A: weights row for this lane's c_out (M = lane%16 in both halves)
        const unsigned short* arow = &wlds[(tap*64 + mbase + nn)*64];
        #pragma unroll
        for (int kc = 0; kc < 2; ++kc) {
            Frag a, bfr;
            // A 16x32 bf16: lanes 0-15 hold K {0..7,16..23}, lanes 16-31 hold K {8..15,24..31}
            a.q[0] = *reinterpret_cast<const uint4*>(arow + kc*32 + hi*8);
            a.q[1] = *reinterpret_cast<const uint4*>(arow + kc*32 + hi*8 + 16);
            // B 32x16 bf16: lanes 0-15 hold K {0..7,16..23}, lanes 16-31 hold K {8..15,24..31}
            const unsigned short* bp = brow + kc*32 + hi*8;
            bfr.q[0] = *reinterpret_cast<const uint4*>(bp);
            bfr.q[1] = *reinterpret_cast<const uint4*>(bp + 16);
            acc = __builtin_amdgcn_wmma_f32_16x16x32_bf16(
                      false, a.v, false, bfr.v, (short)0, acc, false, false);
        }
    }

    // Fused LIF epilogue: D layout VGPR r -> M = r + 8*hi, N = lane&15.
    #pragma unroll
    for (int r = 0; r < 8; ++r) {
        int c  = mbase + r + hi*8;
        int px = x0 + nn;
        int g  = (((b*64 + c)*64) + y)*64 + px;
        float mem   = membrane[g] * 0.99f + acc[r];
        float th    = thresholds[g];
        bool  fired = mem > th;
        float spike = fired ? 1.0f : 0.0f;
        out_spk[g] = spike;
        out_mem[g] = fired ? 0.1f : mem;
        out_th[g]  = fired ? 1.0f : th * 0.95f;
        float tp = trace_post[g] * 0.95f + spike;
        out_post[g] = clip01(tp);
    }
}

// ---------------------------------------------------------------------------
// Kernel 2: new_pre = clip(trace_pre*0.95 + in_spikes, 0, 1)  (float4 streams)
// ---------------------------------------------------------------------------
__global__ __launch_bounds__(256) void snn_pre_kernel(
    const float4* __restrict__ in_spikes, const float4* __restrict__ trace_pre,
    float4* __restrict__ out_pre)
{
    int g = blockIdx.x * 256 + threadIdx.x;   // over NTOT/4
    float4 s = in_spikes[g], t = trace_pre[g], r;
    r.x = clip01(t.x * 0.95f + s.x);
    r.y = clip01(t.y * 0.95f + s.y);
    r.z = clip01(t.z * 0.95f + s.z);
    r.w = clip01(t.w * 0.95f + s.w);
    out_pre[g] = r;
}

// ---------------------------------------------------------------------------
// Kernel 3: batch means avg_pre/avg_post (coalesced float4 column reduction)
// ---------------------------------------------------------------------------
__global__ __launch_bounds__(256) void snn_avg_kernel(
    const float4* __restrict__ new_pre, const float4* __restrict__ new_post,
    float4* __restrict__ avg_pre, float4* __restrict__ avg_post)
{
    int j = blockIdx.x * 256 + threadIdx.x;   // over CHW/4
    float4 s0 = {0,0,0,0}, s1 = {0,0,0,0};
    for (int bb = 0; bb < 64; ++bb) {
        float4 a = new_pre [bb*(CHW/4) + j];
        float4 c = new_post[bb*(CHW/4) + j];
        s0.x += a.x; s0.y += a.y; s0.z += a.z; s0.w += a.w;
        s1.x += c.x; s1.y += c.y; s1.z += c.z; s1.w += c.w;
    }
    const float inv = 1.0f / 64.0f;
    s0.x *= inv; s0.y *= inv; s0.z *= inv; s0.w *= inv;
    s1.x *= inv; s1.y *= inv; s1.z *= inv; s1.w *= inv;
    avg_pre[j] = s0; avg_post[j] = s1;
}

// ---------------------------------------------------------------------------
// Kernel 4: delta[c,i,j] = corr(avg_post[c], A_POS*avg_pre[c]-A_NEG*avg_post[c])
// (mean_b of pot-dep collapses to this since conv is linear in new_post)
// then new_w = clip(weights + delta, +-0.1). One block per out-channel.
// ---------------------------------------------------------------------------
__global__ __launch_bounds__(256) void snn_stdp_kernel(
    const float* __restrict__ avg_pre, const float* __restrict__ avg_post,
    const float* __restrict__ weights, float* __restrict__ out_w)
{
    __shared__ float red[9*256];
    const int c = blockIdx.x, t = threadIdx.x;
    const float* ap = avg_pre  + c*4096;
    const float* aq = avg_post + c*4096;

    float part[9];
    #pragma unroll
    for (int k = 0; k < 9; ++k) part[k] = 0.0f;

    for (int p = t; p < 4096; p += 256) {
        int h = p >> 6, w = p & 63;
        float kd = 0.005f*ap[p] - 0.005f*aq[p];   // A_POS*avg_pre - A_NEG*avg_post
        #pragma unroll
        for (int i = 0; i < 3; ++i) {
            #pragma unroll
            for (int j = 0; j < 3; ++j) {
                int yy = h + i - 1, xx = w + j - 1;
                if ((unsigned)yy < 64u && (unsigned)xx < 64u)
                    part[i*3 + j] += aq[yy*64 + xx] * kd;
            }
        }
    }
    #pragma unroll
    for (int k = 0; k < 9; ++k) red[k*256 + t] = part[k];
    __syncthreads();
    for (int s = 128; s > 0; s >>= 1) {
        if (t < s) {
            #pragma unroll
            for (int k = 0; k < 9; ++k) red[k*256 + t] += red[k*256 + t + s];
        }
        __syncthreads();
    }
    // broadcast delta[c,*] and write clipped weights for this out-channel
    for (int idx = t; idx < 64*9; idx += 256) {
        int ci = idx / 9, tap = idx % 9;
        float w = weights[(c*64 + ci)*9 + tap] + red[tap*256];
        out_w[(c*64 + ci)*9 + tap] = fminf(fmaxf(w, -0.1f), 0.1f);
    }
}

extern "C" void kernel_launch(void* const* d_in, const int* in_sizes, int n_in,
                              void* d_out, int out_size, void* d_ws, size_t ws_size,
                              hipStream_t stream) {
    (void)in_sizes; (void)n_in; (void)out_size; (void)ws_size;
    const float* in_spikes  = (const float*)d_in[0];
    const float* weights    = (const float*)d_in[1];
    const float* membrane   = (const float*)d_in[2];
    const float* thresholds = (const float*)d_in[3];
    const float* trace_pre  = (const float*)d_in[4];
    const float* trace_post = (const float*)d_in[5];

    float* out      = (float*)d_out;
    float* out_spk  = out;
    float* out_mem  = out + (size_t)1*NTOT;
    float* out_th   = out + (size_t)2*NTOT;
    float* out_pre  = out + (size_t)3*NTOT;
    float* out_post = out + (size_t)4*NTOT;
    float* out_w    = out + (size_t)5*NTOT;

    float*          avg_pre  = (float*)d_ws;
    float*          avg_post = avg_pre + CHW;
    unsigned short* wbf      = (unsigned short*)(avg_post + CHW);

    snn_wprep_kernel<<<NW/256, 256, 0, stream>>>(weights, wbf);
    snn_pre_kernel<<<NTOT/1024, 256, 0, stream>>>(
        (const float4*)in_spikes, (const float4*)trace_pre, (float4*)out_pre);
    snn_conv_lif_kernel<<<64*64, 512, 0, stream>>>(
        in_spikes, wbf, membrane, thresholds, trace_post,
        out_spk, out_mem, out_th, out_post);
    snn_avg_kernel<<<CHW/1024, 256, 0, stream>>>(
        (const float4*)out_pre, (const float4*)out_post,
        (float4*)avg_pre, (float4*)avg_post);
    snn_stdp_kernel<<<64, 256, 0, stream>>>(avg_pre, avg_post, weights, out_w);
}
